// GraphAttentionLayer_35510789603493
// MI455X (gfx1250) — compile-verified, matrix operand order checked by reading, not run on hardware
//
#include <hip/hip_runtime.h>

typedef __attribute__((ext_vector_type(16))) _Float16 v16h;
typedef __attribute__((ext_vector_type(8)))  _Float16 v8h;
typedef __attribute__((ext_vector_type(8)))  float    v8f;

#define BB 16
#define NN 1024
#define FIN 64
#define FOUT 64
#define HH 4
#define NEG_SLOPE 0.2f

__device__ __forceinline__ float lrelu(float e) {
    return e > 0.f ? e : NEG_SLOPE * e;
}

// Build a v16h operand from two contiguous 8-half (16B) chunks.
// Half->K map for v_wmma_*_16x16x32_f16 (ISA 7.12.2): halves 0..7 cover
// K = 8*hi + {0..7}, halves 8..15 cover K = 16 + 8*hi + {0..7}.
__device__ __forceinline__ v16h load_b_operand(const _Float16* base, int hi) {
    const v8h lo = *(const v8h*)(base + 8 * hi);
    const v8h hh = *(const v8h*)(base + 16 + 8 * hi);
    v16h r;
    for (int i = 0; i < 8; ++i) { r[i] = lo[i]; r[8 + i] = hh[i]; }
    return r;
}

// Build the 16 attention weights this lane contributes to an A operand,
// from two contiguous 8-float runs of adj / dst.
__device__ __forceinline__ v16h make_p_tile(const float* arow, const float* dsh,
                                            int jb1, int jb2, float srow,
                                            float emax, float& psum) {
    float4 a0 = *(const float4*)(arow + jb1);
    float4 a1 = *(const float4*)(arow + jb1 + 4);
    float4 a2 = *(const float4*)(arow + jb2);
    float4 a3 = *(const float4*)(arow + jb2 + 4);
    float4 d0 = *(const float4*)&dsh[jb1];
    float4 d1 = *(const float4*)&dsh[jb1 + 4];
    float4 d2 = *(const float4*)&dsh[jb2];
    float4 d3 = *(const float4*)&dsh[jb2 + 4];

    float av[16] = {a0.x,a0.y,a0.z,a0.w, a1.x,a1.y,a1.z,a1.w,
                    a2.x,a2.y,a2.z,a2.w, a3.x,a3.y,a3.z,a3.w};
    float dv[16] = {d0.x,d0.y,d0.z,d0.w, d1.x,d1.y,d1.z,d1.w,
                    d2.x,d2.y,d2.z,d2.w, d3.x,d3.y,d3.z,d3.w};
    v16h pa;
    for (int t = 0; t < 16; ++t) {
        float p = 0.f;
        if (av[t] > 0.f)
            p = __expf(lrelu(srow + dv[t]) - emax);
        psum += p;
        pa[t] = (_Float16)p;
    }
    return pa;
}

// Kernel 0: Wt16[h][o][k] = (f16) W[h][k][o]  (32 KB, one-shot)
__global__ void __launch_bounds__(256)
gat_transpose_w_kernel(const float* __restrict__ W, _Float16* __restrict__ Wt16) {
    int idx = blockIdx.x * 256 + threadIdx.x;      // HH*FOUT*FIN = 16384
    int k = idx & 63;
    int o = (idx >> 6) & 63;
    int h = idx >> 12;
    Wt16[idx] = (_Float16)W[((size_t)h * FIN + k) * FOUT + o];
}

// Kernel 1: Wh = x @ W per (b,h) via WMMA; emit Wh TRANSPOSED ([o][j], f16)
// so kernel 2's B operands are contiguous b128 loads. Also src/dst rows.
__global__ void __launch_bounds__(32)
gat_project_kernel(const float* __restrict__ x,
                   const _Float16* __restrict__ Wt16,
                   const float* __restrict__ a_src,
                   const float* __restrict__ a_dst,
                   _Float16* __restrict__ WhT,
                   float* __restrict__ srcv,
                   float* __restrict__ dstv)
{
    __shared__ float whtile[16 * FOUT];

    const int wg = blockIdx.x;               // B*H*(N/16) blocks
    const int mt = wg % (NN / 16);
    const int bh = wg / (NN / 16);
    const int h  = bh % HH;
    const int b  = bh / HH;
    const int lane = threadIdx.x;
    const int hi = lane >> 4;
    const int m  = lane & 15;
    const int i0 = mt * 16;

    // A operand: x rows, two contiguous float8 runs per k-chunk (b128 pairs)
    v16h a[2];
    const float* xrow = x + ((size_t)b * NN + (i0 + m)) * FIN;
    for (int kc = 0; kc < 2; ++kc) {
        const float4* p1 = (const float4*)(xrow + kc * 32 + 8 * hi);
        const float4* p2 = (const float4*)(xrow + kc * 32 + 16 + 8 * hi);
        float4 x0 = p1[0], x1 = p1[1], x2 = p2[0], x3 = p2[1];
        a[kc][0]  = (_Float16)x0.x; a[kc][1]  = (_Float16)x0.y;
        a[kc][2]  = (_Float16)x0.z; a[kc][3]  = (_Float16)x0.w;
        a[kc][4]  = (_Float16)x1.x; a[kc][5]  = (_Float16)x1.y;
        a[kc][6]  = (_Float16)x1.z; a[kc][7]  = (_Float16)x1.w;
        a[kc][8]  = (_Float16)x2.x; a[kc][9]  = (_Float16)x2.y;
        a[kc][10] = (_Float16)x2.z; a[kc][11] = (_Float16)x2.w;
        a[kc][12] = (_Float16)x3.x; a[kc][13] = (_Float16)x3.y;
        a[kc][14] = (_Float16)x3.z; a[kc][15] = (_Float16)x3.w;
    }

    for (int ot = 0; ot < 4; ++ot) {
        v8f c = {};
        const int o = ot * 16 + m;
        const _Float16* wcol = Wt16 + ((size_t)h * FOUT + o) * FIN;
        for (int kc = 0; kc < 2; ++kc) {
            v16h bm = load_b_operand(wcol + kc * 32, hi);
            c = __builtin_amdgcn_wmma_f32_16x16x32_f16(
                    false, a[kc], false, bm, (short)0, c, false, false);
        }
        for (int v = 0; v < 8; ++v)
            whtile[(v + 8 * hi) * FOUT + o] = c[v];
    }
    __syncthreads();

    // Transposed spill: WhT[(bh*FOUT + o)*N + j], 16B vector stores
    for (int oo = lane; oo < FOUT; oo += 32) {
        v8h t0, t1;
        for (int r = 0; r < 8; ++r) {
            t0[r] = (_Float16)whtile[r * FOUT + oo];
            t1[r] = (_Float16)whtile[(r + 8) * FOUT + oo];
        }
        _Float16* dp = WhT + ((size_t)bh * FOUT + oo) * NN + i0;
        *(v8h*)dp       = t0;
        *(v8h*)(dp + 8) = t1;
    }

    if (lane < 16) {
        float s = 0.f, d = 0.f;
        for (int o = 0; o < FOUT; ++o) {
            float w = whtile[lane * FOUT + o];
            s += w * a_src[h * FOUT + o];
            d += w * a_dst[h * FOUT + o];
        }
        size_t ridx = (size_t)bh * NN + i0 + lane;
        srcv[ridx] = s;
        dstv[ridx] = d;
    }
}

// Kernel 2: fused masked softmax + aggregation. One wave per 32-row block
// (two 16-row M-tiles sharing every B-operand load -> 8 WMMAs per 8 b128
// loads, halving WhT L2 traffic). Exact row max via max_{adj} dst_j
// (leaky_relu monotone) => no online rescaling of WMMA accumulators.
__global__ void __launch_bounds__(32)
gat_attention_kernel(const float* __restrict__ adj,
                     const _Float16* __restrict__ WhT,
                     const float* __restrict__ srcv,
                     const float* __restrict__ dstv,
                     float* __restrict__ out)
{
    __shared__ float dsh[NN];
    __shared__ float msh[32];
    __shared__ float ssum[32];
    __shared__ float srcsh[32];

    const int wg = blockIdx.x;               // B*H*(N/32) blocks
    const int mt = wg % (NN / 32);
    const int bh = wg / (NN / 32);
    const int h  = bh % HH;
    const int b  = bh / HH;
    const int lane = threadIdx.x;
    const int hi = lane >> 4;
    const int m  = lane & 15;
    const int i0 = mt * 32;

    const size_t bhN = (size_t)bh * NN;

    for (int j4 = lane * 4; j4 < NN; j4 += 128)
        *(float4*)&dsh[j4] = *(const float4*)(dstv + bhN + j4);
    srcsh[lane] = srcv[bhN + i0 + lane];
    __syncthreads();

    // Phase A: per-row max of dst over neighbors (vectorized scan, 32 rows)
    for (int r = 0; r < 32; ++r) {
        const float* arow = adj + (size_t)(i0 + r) * NN;
        float mx = -3.4e38f;
        for (int j4 = lane * 4; j4 < NN; j4 += 128) {
            float4 av = *(const float4*)(arow + j4);
            float4 dv = *(const float4*)&dsh[j4];
            if (av.x > 0.f) mx = fmaxf(mx, dv.x);
            if (av.y > 0.f) mx = fmaxf(mx, dv.y);
            if (av.z > 0.f) mx = fmaxf(mx, dv.z);
            if (av.w > 0.f) mx = fmaxf(mx, dv.w);
        }
        for (int off = 16; off > 0; off >>= 1)
            mx = fmaxf(mx, __shfl_xor(mx, off, 32));
        if (lane == 0) msh[r] = mx;
    }
    __syncthreads();

    // Phase B: stream j in chunks of 32; two P tiles share each B operand
    v8f zero = {};
    v8f c0[4] = {zero, zero, zero, zero};
    v8f c1[4] = {zero, zero, zero, zero};
    const float srow0 = srcsh[m];
    const float srow1 = srcsh[16 + m];
    const float emax0 = lrelu(srow0 + msh[m]);
    const float emax1 = lrelu(srow1 + msh[16 + m]);
    const float* arow0 = adj + (size_t)(i0 + m) * NN;
    const float* arow1 = adj + (size_t)(i0 + 16 + m) * NN;
    float psum0 = 0.f, psum1 = 0.f;

    for (int j0 = 0; j0 < NN; j0 += 32) {
        if (j0 + 32 < NN) {
            __builtin_prefetch(arow0 + j0 + 32, 0, 0);  // global_prefetch_b8
            __builtin_prefetch(arow1 + j0 + 32, 0, 0);
        }
        const int jb1 = j0 + 8 * hi;        // halves 0..7
        const int jb2 = j0 + 16 + 8 * hi;   // halves 8..15

        v16h pa0 = make_p_tile(arow0, dsh, jb1, jb2, srow0, emax0, psum0);
        v16h pa1 = make_p_tile(arow1, dsh, jb1, jb2, srow1, emax1, psum1);

        // B operands loaded ONCE, consumed by both M-tiles
        for (int ot = 0; ot < 4; ++ot) {
            const int o = ot * 16 + m;
            const _Float16* wcol = WhT + ((size_t)bh * FOUT + o) * NN + j0;
            v16h bm = load_b_operand(wcol, hi);
            c0[ot] = __builtin_amdgcn_wmma_f32_16x16x32_f16(
                         false, pa0, false, bm, (short)0, c0[ot], false, false);
            c1[ot] = __builtin_amdgcn_wmma_f32_16x16x32_f16(
                         false, pa1, false, bm, (short)0, c1[ot], false, false);
        }
    }

    // Combine lane pairs (m, m+16) holding the same row, publish denominators
    psum0 += __shfl_xor(psum0, 16, 32);
    psum1 += __shfl_xor(psum1, 16, 32);
    if (lane < 16) { ssum[lane] = psum0; ssum[16 + lane] = psum1; }
    __syncthreads();

    // Phase C: normalize, scatter into head-concatenated output
    for (int ot = 0; ot < 4; ++ot) {
        const int o = ot * 16 + m;
        for (int v = 0; v < 8; ++v) {
            int row = v + 8 * hi;
            float v0 = c0[ot][v] / ssum[row];
            float v1 = c1[ot][v] / ssum[16 + row];
            out[((size_t)b * NN + (i0 + row)) * (HH * FOUT) + h * FOUT + o] = v0;
            out[((size_t)b * NN + (i0 + 16 + row)) * (HH * FOUT) + h * FOUT + o] = v1;
        }
    }
}

extern "C" void kernel_launch(void* const* d_in, const int* in_sizes, int n_in,
                              void* d_out, int out_size, void* d_ws, size_t ws_size,
                              hipStream_t stream) {
    (void)in_sizes; (void)n_in; (void)out_size; (void)ws_size;
    const float* x     = (const float*)d_in[0];
    const float* adj   = (const float*)d_in[1];
    const float* W     = (const float*)d_in[2];
    const float* a_src = (const float*)d_in[3];
    const float* a_dst = (const float*)d_in[4];
    float* out = (float*)d_out;

    // Workspace layout (16B-aligned slabs)
    char* ws = (char*)d_ws;
    _Float16* WhT = (_Float16*)ws;                              // 8 MB
    size_t off = (size_t)BB * HH * FOUT * NN * sizeof(_Float16);
    float* srcv = (float*)(ws + off);  off += (size_t)BB * HH * NN * sizeof(float);
    float* dstv = (float*)(ws + off);  off += (size_t)BB * HH * NN * sizeof(float);
    _Float16* Wt16 = (_Float16*)(ws + off);                     // 32 KB

    gat_transpose_w_kernel<<<(HH * FIN * FOUT) / 256, 256, 0, stream>>>(W, Wt16);

    gat_project_kernel<<<BB * HH * (NN / 16), 32, 0, stream>>>(
        x, Wt16, a_src, a_dst, WhT, srcv, dstv);
    gat_attention_kernel<<<BB * HH * (NN / 32), 32, 0, stream>>>(
        adj, WhT, srcv, dstv, out);
}
